// pi_posterior_module_88776974008911
// MI455X (gfx1250) — matrix operand order repaired and checked by reading, not compiled, not run on hardware
//
#include <hip/hip_runtime.h>
#include <hip/hip_bf16.h>

typedef __attribute__((ext_vector_type(16))) _Float16 v16h;
typedef __attribute__((ext_vector_type(8)))  float    v8f;

#define N_PTS  131072
#define DIM    128
#define KCODES 1024
#define NTILE  (KCODES / 16)   // 64 column tiles of 16 codes
#define NCHUNK (DIM / 32)      // 4 K-chunks of 32 for wmma f16
#define SMEM_BYTES (NTILE * NCHUNK * 32 * 16 * 2)  // 262144 = 256KB B fragments

// ---------------------------------------------------------------------------
// Kernel 1: codebook prep.
//   e2[k] = ||emb_k||^2  (fp32)
//   embB  = emb converted to f16 in WMMA B-fragment order:
//           fragment (j, c) for column tile j (16 codes) and K-chunk c (32 dims)
//           lane L holds column N = j*16 + (L&15), halves h=0..15 hold
//           K = c*32 + (L>=16 ? 16 : 0) + h.
// ---------------------------------------------------------------------------
__global__ void vq_prep_kernel(const float* __restrict__ emb,
                               _Float16* __restrict__ embB,
                               float* __restrict__ e2) {
    const int t = blockIdx.x * blockDim.x + threadIdx.x;  // 0..8191
    if (t < KCODES) {
        const float* er = emb + (size_t)t * DIM;
        float s = 0.f;
        #pragma unroll 8
        for (int d = 0; d < DIM; ++d) s += er[d] * er[d];
        e2[t] = s;
    }
    if (t < NTILE * NCHUNK * 32) {
        const int lane = t & 31;
        const int c    = (t >> 5) & (NCHUNK - 1);
        const int j    = t >> 7;
        const int nn   = j * 16 + (lane & 15);
        const int kb   = c * 32 + ((lane >> 4) ? 16 : 0);
        __align__(16) _Float16 frag[16];
        #pragma unroll
        for (int h = 0; h < 16; ++h)
            frag[h] = (_Float16)emb[(size_t)nn * DIM + kb + h];
        uint4* dst = (uint4*)(embB + (size_t)t * 16);
        dst[0] = ((const uint4*)frag)[0];
        dst[1] = ((const uint4*)frag)[1];
    }
}

// ---------------------------------------------------------------------------
// Kernel 2: WMMA nearest-code search.
//   One wave owns 16 rows of x.  argmin_k ||x-e_k||^2 == argmin_k (e2[k]-2 x.e_k)
//   Full f16 codebook staged in LDS (256KB) as ready-to-load B fragments.
// ---------------------------------------------------------------------------
__global__ void vq_argmin_kernel(const float* __restrict__ x,
                                 const _Float16* __restrict__ embB,
                                 const float* __restrict__ e2,
                                 int* __restrict__ idx_out,
                                 float* __restrict__ out_index_f) {
    extern __shared__ _Float16 sB[];  // 131072 halves

    // Cooperative copy of B fragments global -> LDS (256 threads x 1KB)
    {
        const uint4* src = (const uint4*)embB;
        uint4*       dst = (uint4*)sB;
        for (int i = threadIdx.x; i < SMEM_BYTES / 16; i += 256) dst[i] = src[i];
    }
    __syncthreads();

    const int wave   = threadIdx.x >> 5;
    const int lane   = threadIdx.x & 31;
    const int m16    = lane & 15;
    const int hi     = lane >> 4;
    const int rowbase = (blockIdx.x * 8 + wave) * 16;
    const int row     = rowbase + m16;

    // Load + convert A fragments (16 rows x 128 dims, fp32 -> f16), contiguous K layout:
    // pair p: K = 2*p + 16*hi  (+ chunk*32)
    v16h a[NCHUNK];
    #pragma unroll
    for (int c = 0; c < NCHUNK; ++c) {
        #pragma unroll
        for (int p = 0; p < 8; ++p) {
            const int k0 = c * 32 + 2 * p + 16 * hi;
            const float2 f = *(const float2*)(x + (size_t)row * DIM + k0);
            a[c][2 * p]     = (_Float16)f.x;
            a[c][2 * p + 1] = (_Float16)f.y;
        }
    }

    float best[8];
    int   bidx[8];
    #pragma unroll
    for (int r = 0; r < 8; ++r) { best[r] = 3.0e38f; bidx[r] = 0; }

    for (int j = 0; j < NTILE; ++j) {
        const v16h* bj = (const v16h*)sB + (size_t)(j * NCHUNK) * 32 + lane;
        v8f acc = {0.f, 0.f, 0.f, 0.f, 0.f, 0.f, 0.f, 0.f};
        #pragma unroll
        for (int c = 0; c < NCHUNK; ++c) {
            const v16h b = bj[c * 32];
            acc = __builtin_amdgcn_wmma_f32_16x16x32_f16(
                false, a[c], false, b, (short)0, acc, false, false);
        }
        const int   col = j * 16 + m16;
        const float ev  = e2[col];
        #pragma unroll
        for (int r = 0; r < 8; ++r) {
            const float v = ev - 2.0f * acc[r];   // row m = r + 8*hi, col = N
            if (v < best[r]) { best[r] = v; bidx[r] = col; }
        }
    }

    // Min-reduce across the 16 lanes holding each row (xor masks stay in-half).
    #pragma unroll
    for (int r = 0; r < 8; ++r) {
        float v = best[r];
        int   i = bidx[r];
        #pragma unroll
        for (int mask = 8; mask >= 1; mask >>= 1) {
            const float ov = __shfl_xor(v, mask, 32);
            const int   oi = __shfl_xor(i, mask, 32);
            if (ov < v || (ov == v && oi < i)) { v = ov; i = oi; }
        }
        bidx[r] = i;
    }
    if (m16 == 0) {  // lanes 0 (rows 0..7) and 16 (rows 8..15)
        #pragma unroll
        for (int r = 0; r < 8; ++r) {
            const int outrow = rowbase + hi * 8 + r;
            idx_out[outrow]     = bidx[r];
            out_index_f[outrow] = (float)bidx[r];
        }
    }
}

// ---------------------------------------------------------------------------
// Kernel 3: exact fp32 gather + straight-through output + loss partials.
//   quantized_st = x + (q - x)   (same op order as the reference)
//   per-block partial of sum((q-x)^2), fixed-order tree -> deterministic.
// ---------------------------------------------------------------------------
__global__ void vq_gather_loss_kernel(const float* __restrict__ x,
                                      const float* __restrict__ emb,
                                      const int* __restrict__ idx,
                                      float* __restrict__ out_q,
                                      float* __restrict__ partial) {
    __shared__ float sred[256];
    const int    t   = blockIdx.x * 256 + threadIdx.x;
    const size_t e   = (size_t)t * 4;
    const int    row = (int)(e >> 7);
    const int    col = (int)(e & 127);
    const int    id  = idx[row];

    const float4 q  = *(const float4*)(emb + (size_t)id * DIM + col);
    const float4 xv = *(const float4*)(x + e);
    float4 d, o;
    d.x = q.x - xv.x; d.y = q.y - xv.y; d.z = q.z - xv.z; d.w = q.w - xv.w;
    o.x = xv.x + d.x; o.y = xv.y + d.y; o.z = xv.z + d.z; o.w = xv.w + d.w;
    *(float4*)(out_q + e) = o;

    sred[threadIdx.x] = d.x * d.x + d.y * d.y + d.z * d.z + d.w * d.w;
    __syncthreads();
    #pragma unroll
    for (int off = 128; off >= 1; off >>= 1) {
        if (threadIdx.x < off) sred[threadIdx.x] += sred[threadIdx.x + off];
        __syncthreads();
    }
    if (threadIdx.x == 0) partial[blockIdx.x] = sred[0];
}

// ---------------------------------------------------------------------------
// Kernel 4: deterministic final reduction.
//   vq_loss = BETA*mean + mean = 1.25 * sum/(N*D)
// ---------------------------------------------------------------------------
__global__ void vq_finalize_kernel(const float* __restrict__ partial,
                                   float* __restrict__ out_loss,
                                   int nparts) {
    __shared__ float sred[256];
    float s = 0.f;
    for (int i = threadIdx.x; i < nparts; i += 256) s += partial[i];
    sred[threadIdx.x] = s;
    __syncthreads();
    #pragma unroll
    for (int off = 128; off >= 1; off >>= 1) {
        if (threadIdx.x < off) sred[threadIdx.x] += sred[threadIdx.x + off];
        __syncthreads();
    }
    if (threadIdx.x == 0)
        out_loss[0] = 1.25f * sred[0] / (float)((size_t)N_PTS * DIM);
}

// ---------------------------------------------------------------------------
extern "C" void kernel_launch(void* const* d_in, const int* in_sizes, int n_in,
                              void* d_out, int out_size, void* d_ws, size_t ws_size,
                              hipStream_t stream) {
    const float* x   = (const float*)d_in[0];   // [N, D] fp32
    const float* emb = (const float*)d_in[1];   // [K, D] fp32
    float*       out = (float*)d_out;           // [N] idx | [N*D] q_st | [1] loss

    char* ws = (char*)d_ws;
    _Float16* embB    = (_Float16*)ws;                               // 256 KB
    float*    e2      = (float*)(ws + 262144);                       //   4 KB
    int*      idxbuf  = (int*)(ws + 262144 + 4096);                  // 512 KB
    float*    partial = (float*)(ws + 262144 + 4096 + 524288);       //  64 KB

    const int gather_blocks = ((size_t)N_PTS * DIM / 4) / 256;       // 16384

    vq_prep_kernel<<<32, 256, 0, stream>>>(emb, embB, e2);
    vq_argmin_kernel<<<N_PTS / (8 * 16), 256, SMEM_BYTES, stream>>>(
        x, embB, e2, idxbuf, out);
    vq_gather_loss_kernel<<<gather_blocks, 256, 0, stream>>>(
        x, emb, idxbuf, out + N_PTS, partial);
    vq_finalize_kernel<<<1, 256, 0, stream>>>(
        partial, out + N_PTS + (size_t)N_PTS * DIM, gather_blocks);
}
